// PCN_35476429866019
// MI455X (gfx1250) — matrix operand order repaired and checked by reading, not compiled, not run on hardware
//
#include <hip/hip_runtime.h>

// ---------------- types matching CDNA5 WMMA operand layouts ----------------
typedef __attribute__((ext_vector_type(16))) __bf16   v16bf;
typedef __attribute__((ext_vector_type(8)))  float    v8f;
typedef __attribute__((ext_vector_type(8)))  unsigned short ushort8;

union FragU { v16bf v; ushort8 h[2]; };

__device__ __forceinline__ unsigned short f2bf(float x) {
    union { float f; unsigned int u; } a; a.f = x;
    unsigned int r = a.u + 0x7FFFu + ((a.u >> 16) & 1u);   // round-to-nearest-even
    return (unsigned short)(r >> 16);
}
__device__ __forceinline__ float bf2f(unsigned short b) {
    union { unsigned int u; float f; } a; a.u = ((unsigned int)b) << 16;
    return a.f;
}

// A-matrix 16x32 bf16 fragment (ISA 7.12.2): lane = h*16+m rows m=0..15,
// half h selects K-subblock; per lane two contiguous 16B chunks:
//   k in [k0+8h, k0+8h+7] and [k0+16+8h, k0+16+8h+7]
__device__ __forceinline__ v16bf load_A(const unsigned short* buf, int stride,
                                        int m0, int k0, int lane) {
    int m = m0 + (lane & 15);
    int h = lane >> 4;
    const unsigned short* p = buf + m * stride + k0 + 8 * h;
    FragU f;
    f.h[0] = *(const ushort8*)(p);
    f.h[1] = *(const ushort8*)(p + 16);
    return f.v;
}

// B-matrix 32x16 bf16 fragment: lane l holds column n=l%16, K in [k0+8h, k0+8h+7] and [k0+16+8h, k0+16+8h+7], h=l/16
// We store every B operand as [n][k] row-major so these are two contiguous 16B runs.
__device__ __forceinline__ v16bf load_B(const unsigned short* wnk, int kstride,
                                        int n0, int k0, int lane) {
    int n = n0 + (lane & 15);
    int h = lane >> 4;
    const unsigned short* p = wnk + n * kstride + k0 + 8 * h;
    FragU f;
    f.h[0] = *(const ushort8*)(p);
    f.h[1] = *(const ushort8*)(p + 16);
    return f.v;
}

#define WMMA_BF16(A, Bf, C) \
    __builtin_amdgcn_wmma_f32_16x16x32_bf16(false, (A), false, (Bf), (short)0, (C), false, false)

// ---------------- problem constants ----------------
#define BB   64
#define H0   784
#define H1   256
#define H2   128
#define H3   64
#define K0P  800          // H0 padded to multiple of 32 for the e0@W1 GEMM
#define LRATE 0.1f

// LDS row strides (ushorts), each row 16B-aligned, +16B pad vs. natural stride
#define A1S  (H1 + 8)     // 264
#define A2S  (H2 + 8)     // 136
#define E0S  (K0P + 8)    // 808
#define E1S  (H1 + 8)     // 264

// LDS byte offsets
#define OFF_S1   0                       // 64*256*4  = 65536
#define OFF_S2   65536                   // 64*128*4  = 32768
#define OFF_A1   98304                   // 64*264*2  = 33792
#define OFF_A2   132096                  // 64*136*2  = 17408
#define OFF_E0   149504                  // 64*808*2  = 103424
#define OFF_E1   252928                  // 64*264*2  = 33792
#define OFF_LAC  286720                  // 16
#define LDS_BYTES 286736                 // < 320KB per workgroup

// ---------------- prep: bf16 weight copies (both orientations), pred2c, s3 copy ----
__global__ void pcn_prep(const float* __restrict__ W1, const float* __restrict__ W2,
                         const float* __restrict__ W3, const float* __restrict__ s3,
                         unsigned short* __restrict__ W1b, unsigned short* __restrict__ T1b,
                         unsigned short* __restrict__ W2b, unsigned short* __restrict__ T2b,
                         float* __restrict__ pred2c, float* __restrict__ outS3) {
    int stride = gridDim.x * blockDim.x;
    int i0 = blockIdx.x * blockDim.x + threadIdx.x;
    // W1b[n][k] = W1[n][k]  (784 x 256, direct bf16 copy; B for pred0)
    for (int i = i0; i < H0 * H1; i += stride) W1b[i] = f2bf(W1[i]);
    // T1b[n][k] = W1[k][n]  (256 x 800, zero-padded; B for d1 = e0 @ W1)
    for (int i = i0; i < H1 * K0P; i += stride) {
        int n = i / K0P, k = i % K0P;
        T1b[i] = (k < H0) ? f2bf(W1[k * H1 + n]) : (unsigned short)0;
    }
    // W2b[n][k] = W2[n][k]  (256 x 128; B for pred1)
    for (int i = i0; i < H1 * H2; i += stride) W2b[i] = f2bf(W2[i]);
    // T2b[n][k] = W2[k][n]  (128 x 256; B for d2 = e1 @ W2)
    for (int i = i0; i < H2 * H1; i += stride) {
        int n = i / H1, k = i % H1;
        T2b[i] = f2bf(W2[k * H2 + n]);
    }
    // pred2c = relu(s3) @ W3^T  (64 x 128), loop-invariant
    for (int i = i0; i < BB * H2; i += stride) {
        int m = i / H2, n = i % H2;
        float s = 0.f;
        for (int k = 0; k < H3; ++k) {
            float v = s3[m * H3 + k];
            v = v > 0.f ? v : 0.f;
            s += v * W3[n * H3 + k];
        }
        pred2c[i] = s;
    }
    // output 0: s_last == s3 (never updated by the reference)
    for (int i = i0; i < BB * H3; i += stride) outS3[i] = s3[i];
}

// ---------------- persistent kernel: 1 workgroup = 1 WGP, 32 wave32 waves ----------
__global__ void __launch_bounds__(1024, 1)
pcn_persistent(const float* __restrict__ input,
               const float* __restrict__ s1_0, const float* __restrict__ s2_0,
               const unsigned short* __restrict__ W1b, const unsigned short* __restrict__ T1b,
               const unsigned short* __restrict__ W2b, const unsigned short* __restrict__ T2b,
               const float* __restrict__ pred2c, const int* __restrict__ cyclesPtr,
               float* __restrict__ outLoss) {
    extern __shared__ char smem[];
    float*          s1s = (float*)(smem + OFF_S1);
    float*          s2s = (float*)(smem + OFF_S2);
    unsigned short* a1s = (unsigned short*)(smem + OFF_A1);
    unsigned short* a2s = (unsigned short*)(smem + OFF_A2);
    unsigned short* e0s = (unsigned short*)(smem + OFF_E0);
    unsigned short* e1s = (unsigned short*)(smem + OFF_E1);
    float*          lac = (float*)(smem + OFF_LAC);

    const int tid  = threadIdx.x;
    const int w    = tid >> 5;          // wave id 0..31
    const int lane = tid & 31;
    const int nl   = lane & 15;         // C/D: n = lane%16
    const int hh   = lane >> 4;         // C/D: m = r + 8*(lane/16)

    // ---- init: states into LDS, zero K-pad of e0, zero loss accumulators ----
    for (int i = tid; i < BB * H1; i += 1024) s1s[i] = s1_0[i];
    for (int i = tid; i < BB * H2; i += 1024) s2s[i] = s2_0[i];
    for (int i = tid; i < BB * (K0P - H0); i += 1024) {
        int m = i / (K0P - H0), k = H0 + i % (K0P - H0);
        e0s[m * E0S + k] = 0;
    }
    if (tid < 4) lac[tid] = 0.f;
    const int cycles = cyclesPtr[0];
    __syncthreads();

    float le0 = 0.f, le1 = 0.f, le2 = 0.f;

    for (int c = 0; c < cycles; ++c) {
        const bool last = (c == cycles - 1);

        // ---- phase A: a1 = relu(s1), a2 = relu(s2) in bf16 ----
        for (int i = tid; i < BB * H1; i += 1024) {
            float v = s1s[i];
            a1s[(i >> 8) * A1S + (i & 255)] = f2bf(v > 0.f ? v : 0.f);
        }
        for (int i = tid; i < BB * H2; i += 1024) {
            float v = s2s[i];
            a2s[(i >> 7) * A2S + (i & 127)] = f2bf(v > 0.f ? v : 0.f);
        }
        __syncthreads();

        // ---- phase B1: pred0 -> e0 (196 tiles), pred1 -> e1 (64 tiles) ----
        for (int t = w; t < 196 + 64; t += 32) {
            v8f acc = {};
            if (t < 196) {                               // e0 = input - relu(s1) @ W1^T
                int m0 = (t / 49) * 16, n0 = (t % 49) * 16;
                for (int kk = 0; kk < H1; kk += 32) {
                    v16bf A  = load_A(a1s, A1S, m0, kk, lane);
                    v16bf Bf = load_B(W1b, H1, n0, kk, lane);
                    acc = WMMA_BF16(A, Bf, acc);
                }
                int n = n0 + nl;
                for (int r = 0; r < 8; ++r) {
                    int m = m0 + r + 8 * hh;
                    float e = input[m * H0 + n] - acc[r];
                    if (last) le0 += e * e;
                    e0s[m * E0S + n] = f2bf(e);
                }
            } else {                                     // e1 = s1 - relu(s2) @ W2^T
                int idx = t - 196;
                int m0 = (idx >> 4) << 4, n0 = (idx & 15) << 4;
                for (int kk = 0; kk < H2; kk += 32) {
                    v16bf A  = load_A(a2s, A2S, m0, kk, lane);
                    v16bf Bf = load_B(W2b, H2, n0, kk, lane);
                    acc = WMMA_BF16(A, Bf, acc);
                }
                int n = n0 + nl;
                for (int r = 0; r < 8; ++r) {
                    int m = m0 + r + 8 * hh;
                    float e = s1s[m * H1 + n] - acc[r];
                    if (last) le1 += e * e;
                    e1s[m * E1S + n] = f2bf(e);
                }
            }
        }
        __syncthreads();

        // ---- phase B2: d1 = e0@W1 -> update s1 (64 tiles); d2 = e1@W2 -> update s2 (32) ----
        for (int t = w; t < 64 + 32; t += 32) {
            v8f acc = {};
            if (t < 64) {
                int m0 = (t >> 4) << 4, n0 = (t & 15) << 4;
                for (int kk = 0; kk < K0P; kk += 32) {   // 25 WMMAs, K zero-padded
                    v16bf A  = load_A(e0s, E0S, m0, kk, lane);
                    v16bf Bf = load_B(T1b, K0P, n0, kk, lane);
                    acc = WMMA_BF16(A, Bf, acc);
                }
                int n = n0 + nl;
                for (int r = 0; r < 8; ++r) {
                    int m = m0 + r + 8 * hh;
                    float sv = s1s[m * H1 + n];
                    float ev = bf2f(e1s[m * E1S + n]);
                    float d  = sv > 0.f ? acc[r] : 0.f;  // relu'(s1) gate
                    s1s[m * H1 + n] = sv + LRATE * (d - ev);
                }
            } else {
                int idx = t - 64;
                int m0 = (idx >> 3) << 4, n0 = (idx & 7) << 4;
                for (int kk = 0; kk < H1; kk += 32) {
                    v16bf A  = load_A(e1s, E1S, m0, kk, lane);
                    v16bf Bf = load_B(T2b, H1, n0, kk, lane);
                    acc = WMMA_BF16(A, Bf, acc);
                }
                int n = n0 + nl;
                for (int r = 0; r < 8; ++r) {
                    int m = m0 + r + 8 * hh;
                    float sv = s2s[m * H2 + n];
                    float e2 = sv - pred2c[m * H2 + n];
                    if (last) le2 += e2 * e2;
                    float d  = sv > 0.f ? acc[r] : 0.f;  // relu'(s2) gate
                    s2s[m * H2 + n] = sv + LRATE * (d - e2);
                }
            }
        }
        __syncthreads();
    }

    // ---- final loss = mean of the three per-error MSEs (ds_add_f32 reduce) ----
    atomicAdd(&lac[0], le0);
    atomicAdd(&lac[1], le1);
    atomicAdd(&lac[2], le2);
    __syncthreads();
    if (tid == 0) {
        float loss = (lac[0] / (float)(BB * H0) +
                      lac[1] / (float)(BB * H1) +
                      lac[2] / (float)(BB * H2)) * (1.0f / 3.0f);
        outLoss[0] = loss;
    }
}

// ---------------- host glue ----------------
extern "C" void kernel_launch(void* const* d_in, const int* in_sizes, int n_in,
                              void* d_out, int out_size, void* d_ws, size_t ws_size,
                              hipStream_t stream) {
    (void)in_sizes; (void)n_in; (void)out_size; (void)ws_size;
    const float* input = (const float*)d_in[0];
    const float* s1    = (const float*)d_in[1];
    const float* s2    = (const float*)d_in[2];
    const float* s3    = (const float*)d_in[3];
    const float* W1    = (const float*)d_in[4];
    const float* W2    = (const float*)d_in[5];
    const float* W3    = (const float*)d_in[6];
    const int*   cyc   = (const int*)d_in[8];
    float*       out   = (float*)d_out;

    // workspace layout (bytes), all 16B-aligned
    char* ws = (char*)d_ws;
    unsigned short* W1b = (unsigned short*)(ws + 0);        // 784*256*2 = 401408
    unsigned short* T1b = (unsigned short*)(ws + 401408);   // 256*800*2 = 409600
    unsigned short* W2b = (unsigned short*)(ws + 811008);   // 256*128*2 =  65536
    unsigned short* T2b = (unsigned short*)(ws + 876544);   // 128*256*2 =  65536
    float*        pred2 = (float*)(ws + 942080);            //  64*128*4 =  32768

    pcn_prep<<<dim3(512), dim3(256), 0, stream>>>(W1, W2, W3, s3,
                                                  W1b, T1b, W2b, T2b,
                                                  pred2, out /* s_last -> out[0:4096] */);

    (void)hipFuncSetAttribute((const void*)pcn_persistent,
                              hipFuncAttributeMaxDynamicSharedMemorySize, LDS_BYTES);

    pcn_persistent<<<dim3(1), dim3(1024), LDS_BYTES, stream>>>(
        input, s1, s2, W1b, T1b, W2b, T2b, pred2, cyc, out + BB * H3 /* loss */);
}
